// GATLayer_72937134620750
// MI455X (gfx1250) — matrix operand order repaired and checked by reading, not compiled, not run on hardware
//
#include <hip/hip_runtime.h>
#include <stdint.h>

// GAT layer for MI455X (gfx1250), round 2.
// bf16 WMMA everywhere, bit-packed adjacency, transposed V for contiguous
// B-fragments, 32-key flash-attention steps (no zero-padded WMMA).

typedef __attribute__((ext_vector_type(16))) __bf16 v16bf;
typedef __attribute__((ext_vector_type(8)))  __bf16 v8bf;
typedef __attribute__((ext_vector_type(8)))  float  v8f;

#define N_NODES 4096
#define IN_DIM  256
#define OUT_DIM 256
#define QK_LD   512            // bf16 row stride of packed q|k
#define HEADS   4
#define HD      64
#define QT      2              // 16-row query tiles per wave
#define ROWS    8192           // B * N

// ---- workspace layout (bytes) ----
#define OFF_HB   0                                        // hB  [8192][256] bf16
#define OFF_WT   (OFF_HB  + (size_t)ROWS * IN_DIM * 2)    // WtB [768][256]  bf16
#define OFF_QK   (OFF_WT  + (size_t)768 * IN_DIM * 2)     // qkB [8192][512] bf16
#define OFF_VT   (OFF_QK  + (size_t)ROWS * QK_LD * 2)     // vT  [2][4][64][4096] bf16
#define OFF_ADJ  (OFF_VT  + (size_t)2 * HEADS * HD * N_NODES * 2) // [4096][128] u32

__device__ __forceinline__ __bf16 f2bf(float x) { return (__bf16)x; }

__device__ __forceinline__ v8f wmma_bf16(v16bf a, v16bf b, v8f c) {
  return __builtin_amdgcn_wmma_f32_16x16x32_bf16(false, a, false, b, (short)0, c,
                                                 false, false);
}
__device__ __forceinline__ v16bf cat8(v8bf lo, v8bf hi) {
  return __builtin_shufflevector(lo, hi, 0, 1, 2, 3, 4, 5, 6, 7,
                                 8, 9, 10, 11, 12, 13, 14, 15);
}

// ---------------------------------------------------------------------------
// Prep kernels: one-time fp32 -> bf16 / bitmask conversions.
// ---------------------------------------------------------------------------
__global__ __launch_bounds__(256) void prep_h_kernel(
    const float* __restrict__ h, __bf16* __restrict__ hB) {
  const size_t i = (size_t)blockIdx.x * blockDim.x + threadIdx.x; // < 2M
  hB[i] = f2bf(h[i]);
}

__global__ __launch_bounds__(256) void prep_wt_kernel(
    const float* __restrict__ W, __bf16* __restrict__ WtB) {
  const size_t i = (size_t)blockIdx.x * blockDim.x + threadIdx.x; // < 768*256
  const int n = (int)(i / IN_DIM), k = (int)(i % IN_DIM);
  WtB[i] = f2bf(W[(size_t)k * (3 * OUT_DIM) + n]);                // W^T
}

__global__ __launch_bounds__(256) void prep_adj_kernel(
    const float* __restrict__ adj, uint32_t* __restrict__ adjBits) {
  const size_t i = (size_t)blockIdx.x * blockDim.x + threadIdx.x; // < 16M
  const uint32_t bits = (uint32_t)__ballot(adj[i] != 0.0f);       // wave32
  if ((threadIdx.x & 31) == 0) adjBits[i >> 5] = bits;
}

// ---------------------------------------------------------------------------
// Kernel 1: qkv = hB @ W (bf16 WMMA, f32 accum). q,k -> qkB bf16 row-major;
// v -> vT bf16 transposed [b][h][feat][node] via a 512 B LDS tile transpose.
// ---------------------------------------------------------------------------
__global__ __launch_bounds__(32) void qkv_gemm_kernel(
    const __bf16* __restrict__ hB, const __bf16* __restrict__ WtB,
    __bf16* __restrict__ qkB, __bf16* __restrict__ vT)
{
  __shared__ __align__(64) __bf16 sT[16][16];   // [featLocal][nodeLocal]

  const int lane = threadIdx.x;
  const int half = lane >> 4, l16 = lane & 15;
  const int m0 = blockIdx.x * 16;               // row tile in [0, 8192)
  const int n0 = blockIdx.y * 16;               // col tile in [0, 768)

  v8f acc = {};
  const __bf16* ap0 = hB  + (size_t)(m0 + l16) * IN_DIM + half * 16;
  const __bf16* bp0 = WtB + (size_t)(n0 + l16) * IN_DIM + half * 16;

  for (int k0 = 0; k0 < IN_DIM; k0 += 32) {
    // A 16x32: lanes 0-15 K 0..15, lanes 16-31 K 16..31
    v16bf a = *(const v16bf*)(ap0 + k0);
    // B 32x16: lanes 0-15 K 0..15, lanes 16-31 K 16..31; contiguous in W^T
    v16bf b = *(const v16bf*)(bp0 + k0);
    acc = wmma_bf16(a, b, acc);
  }

  if (n0 < 2 * OUT_DIM) {
    // q|k tile: row-major bf16, stride 512
#pragma unroll
    for (int j = 0; j < 8; ++j)
      qkB[(size_t)(m0 + half * 8 + j) * QK_LD + n0 + l16] = f2bf(acc[j]);
  } else {
    // v tile: transpose via LDS, write 16 B rows of vT[b][h][feat][node]
#pragma unroll
    for (int j = 0; j < 8; ++j) sT[l16][half * 8 + j] = f2bf(acc[j]);
    asm volatile("s_wait_dscnt 0x0" ::: "memory");
    const int featLocal = lane >> 1, moff = (lane & 1) * 8;
    const v8bf vals = *(const v8bf*)&sT[featLocal][moff];
    const int nv0 = n0 - 2 * OUT_DIM;                  // 0..255
    const int hh = nv0 >> 6, fh = (nv0 & 63) + featLocal;
    const int b = m0 >> 12, node = (m0 & (N_NODES - 1)) + moff;
    *(v8bf*)(vT + (((size_t)b * HEADS + hh) * HD + fh) * N_NODES + node) = vals;
  }
}

// ---------------------------------------------------------------------------
// Kernel 2: masked flash-attention, 32-key steps. Per step & query tile:
// 4 WMMAs QK^T (2 key sub-tiles x 2 feature chunks), bitmask + online softmax,
// 4 WMMAs P@V with a full K=32 A-matrix (no padding).
// ---------------------------------------------------------------------------
__global__ __launch_bounds__(32) void gat_attn_kernel(
    const __bf16* __restrict__ qkB, const __bf16* __restrict__ vT,
    const uint32_t* __restrict__ adjBits, float* __restrict__ out)
{
  __shared__ __align__(64) __bf16 sP[QT][16][32];   // P tiles, bf16

  const int lane = threadIdx.x;
  const int half = lane >> 4, l16 = lane & 15;
  const int bh = blockIdx.y, b = bh >> 2, hh = bh & (HEADS - 1);
  const int q0 = blockIdx.x * (16 * QT);

  const __bf16* qrow0 = qkB + (size_t)b * N_NODES * QK_LD + hh * HD;
  const __bf16* krow0 = qrow0 + OUT_DIM;
  const __bf16* vtb   = vT + ((size_t)b * HEADS + hh) * HD * N_NODES;

  // Q as bf16 A-matrices (QT tiles x 2 feature chunks), resident in VGPRs.
  v16bf qa[QT][2];
#pragma unroll
  for (int t = 0; t < QT; ++t) {
    const __bf16* qp = qrow0 + (size_t)(q0 + t * 16 + l16) * QK_LD;
#pragma unroll
    for (int c = 0; c < 2; ++c) {
      const int ko = c * 32 + half * 16;
      qa[t][c] = *(const v16bf*)(qp + ko);
    }
  }

  v8f   O[QT][4];
  float m_run[QT][8], l_run[QT][8];
#pragma unroll
  for (int t = 0; t < QT; ++t) {
#pragma unroll
    for (int nt = 0; nt < 4; ++nt) O[t][nt] = {};
#pragma unroll
    for (int j = 0; j < 8; ++j) { m_run[t][j] = -1e30f; l_run[t][j] = 0.0f; }
  }

  for (int kt = 0; kt < N_NODES; kt += 32) {
    // K^T B-fragments: [key sub-tile][feature chunk]; contiguous 32 B loads.
    v16bf kb[2][2];
#pragma unroll
    for (int ks = 0; ks < 2; ++ks) {
      const __bf16* kp = krow0 + (size_t)(kt + ks * 16 + l16) * QK_LD;
      if (kt + 32 < N_NODES)
        __builtin_prefetch(kp + 32 * QK_LD, 0, 1);   // global_prefetch_b8
#pragma unroll
      for (int c = 0; c < 2; ++c) kb[ks][c] = *(const v16bf*)(kp + c * 32 + half * 16);
    }
    // V B-fragments from transposed vT: lane = feature col, 16 contiguous keys.
    v16bf vb[4];
#pragma unroll
    for (int nt = 0; nt < 4; ++nt)
      vb[nt] = *(const v16bf*)(vtb + (size_t)(nt * 16 + l16) * N_NODES + kt + half * 16);

    // Scores + bitmask + online softmax per query tile.
#pragma unroll
    for (int t = 0; t < QT; ++t) {
      v8f s0 = {}, s1 = {};
      s0 = wmma_bf16(qa[t][0], kb[0][0], s0);
      s0 = wmma_bf16(qa[t][1], kb[0][1], s0);
      s1 = wmma_bf16(qa[t][0], kb[1][0], s1);
      s1 = wmma_bf16(qa[t][1], kb[1][1], s1);
#pragma unroll
      for (int j = 0; j < 8; ++j) {
        const int qrow = q0 + t * 16 + half * 8 + j;        // C-layout row
        const uint32_t w = adjBits[(size_t)qrow * (N_NODES / 32) + (kt >> 5)];
        const bool e0 = (w >> l16) & 1u;                    // keys kt+0..15
        const bool e1 = (w >> (16 + l16)) & 1u;             // keys kt+16..31
        const float sc0 = s0[j] * 0.125f, sc1 = s1[j] * 0.125f;
        float mx = fmaxf(e0 ? sc0 : -1e30f, e1 ? sc1 : -1e30f);
#pragma unroll
        for (int d = 1; d < 16; d <<= 1) mx = fmaxf(mx, __shfl_xor(mx, d));
        const float mn   = fmaxf(m_run[t][j], mx);
        const float corr = __expf(m_run[t][j] - mn);
        const float p0 = e0 ? __expf(sc0 - mn) : 0.0f;
        const float p1 = e1 ? __expf(sc1 - mn) : 0.0f;
        float rs = p0 + p1;
#pragma unroll
        for (int d = 1; d < 16; d <<= 1) rs += __shfl_xor(rs, d);
        l_run[t][j] = l_run[t][j] * corr + rs;
        m_run[t][j] = mn;
#pragma unroll
        for (int nt = 0; nt < 4; ++nt) O[t][nt][j] *= corr;
        sP[t][half * 8 + j][l16]      = f2bf(p0);
        sP[t][half * 8 + j][16 + l16] = f2bf(p1);
      }
    }

    asm volatile("s_wait_dscnt 0x0" ::: "memory");  // LDS write->read, same wave

    // P (16x32 bf16, full K) @ V tiles.
#pragma unroll
    for (int t = 0; t < QT; ++t) {
      const __bf16* pr = &sP[t][l16][0];
      v16bf pa = *(const v16bf*)(pr + half * 16);
#pragma unroll
      for (int nt = 0; nt < 4; ++nt) O[t][nt] = wmma_bf16(pa, vb[nt], O[t][nt]);
    }
  }

  // Epilogue: out[b, row, hh*64 + col] = O / l  (fp32)
  float* ob = out + (size_t)b * N_NODES * OUT_DIM + hh * HD;
#pragma unroll
  for (int t = 0; t < QT; ++t) {
#pragma unroll
    for (int j = 0; j < 8; ++j) {
      const int   row = q0 + t * 16 + half * 8 + j;
      const float inv = 1.0f / l_run[t][j];
#pragma unroll
      for (int nt = 0; nt < 4; ++nt)
        ob[(size_t)row * OUT_DIM + nt * 16 + l16] = O[t][nt][j] * inv;
    }
  }
}

// ---------------------------------------------------------------------------
extern "C" void kernel_launch(void* const* d_in, const int* in_sizes, int n_in,
                              void* d_out, int out_size, void* d_ws, size_t ws_size,
                              hipStream_t stream) {
  const float* h   = (const float*)d_in[0];   // [2, 4096, 256]
  const float* W   = (const float*)d_in[1];   // [256, 768]
  const float* adj = (const float*)d_in[2];   // [4096, 4096]
  float* out = (float*)d_out;                 // [2, 4096, 256]

  char* ws = (char*)d_ws;
  __bf16*   hB      = (__bf16*)(ws + OFF_HB);
  __bf16*   WtB     = (__bf16*)(ws + OFF_WT);
  __bf16*   qkB     = (__bf16*)(ws + OFF_QK);
  __bf16*   vT      = (__bf16*)(ws + OFF_VT);
  uint32_t* adjBits = (uint32_t*)(ws + OFF_ADJ);

  (void)in_sizes; (void)n_in; (void)out_size; (void)ws_size;

  prep_h_kernel  <<<(ROWS * IN_DIM) / 256, 256, 0, stream>>>(h, hB);
  prep_wt_kernel <<<(768 * IN_DIM) / 256, 256, 0, stream>>>(W, WtB);
  prep_adj_kernel<<<((size_t)N_NODES * N_NODES) / 256, 256, 0, stream>>>(adj, adjBits);

  dim3 g1(ROWS / 16, (3 * OUT_DIM) / 16);     // 512 x 48 waves
  qkv_gemm_kernel<<<g1, 32, 0, stream>>>(hB, WtB, qkB, vT);

  dim3 g2(N_NODES / (16 * QT), 2 * HEADS);    // 128 x 8 waves
  gat_attn_kernel<<<g2, 32, 0, stream>>>(qkB, vT, adjBits, out);
}
